// RNNSeqMaskEncoder_42709154791831
// MI455X (gfx1250) — matrix operand order repaired and checked by reading, not compiled, not run on hardware
//
#include <hip/hip_runtime.h>
#include <hip/hip_bf16.h>

// ---------------------------------------------------------------------------
// Problem constants (B=16, T=2048, D=512, H=512)
// ---------------------------------------------------------------------------
#define BN   16
#define TN   2048
#define DN   512
#define HN   512
#define G3   (3 * HN)          // 1536 gate rows
#define MTOT (BN * TN)         // 32768 rows of x_proj GEMM

typedef __attribute__((ext_vector_type(16))) __bf16 v16bf;
typedef __attribute__((ext_vector_type(8)))  float  v8f;

union BF16Frag {
    v16bf          v;
    unsigned short s[16];
    uint4          q[2];
};

// Hardware bf16 conversion (v_cvt_bf16_f32) instead of bit-twiddling VALU soup.
__device__ __forceinline__ unsigned short f2bf(float f) {
    __bf16 b = (__bf16)f;
    return __builtin_bit_cast(unsigned short, b);
}
__device__ __forceinline__ float bf2f(unsigned short s) {
    union { unsigned u; float f; } x; x.u = ((unsigned)s) << 16;
    return x.f;
}

// ---------------------------------------------------------------------------
// Kernel 0: deterministic workspace init (h0 = 0, barrier counters = 0)
// ---------------------------------------------------------------------------
__global__ __launch_bounds__(256) void init_ws_kernel(unsigned int* hbuf_u32,
                                                      int* cnt) {
    int i = blockIdx.x * 256 + threadIdx.x;
    if (i < 2 * BN * HN / 2) hbuf_u32[i] = 0u;   // 2 x 16 x 512 bf16
    if (i < TN)              cnt[i]      = 0;
}

// ---------------------------------------------------------------------------
// Kernel A: fp32 -> bf16 bulk conversion (one pass; hot loops then load bf16)
// ---------------------------------------------------------------------------
__global__ __launch_bounds__(256) void cvt_bf16_kernel(
    const float* __restrict__ src, unsigned short* __restrict__ dst, int n4) {
    int i = blockIdx.x * 256 + threadIdx.x;
    if (i >= n4) return;
    float4 f = ((const float4*)src)[i];
    union { unsigned short s[4]; uint2 u; } o;
    o.s[0] = f2bf(f.x); o.s[1] = f2bf(f.y);
    o.s[2] = f2bf(f.z); o.s[3] = f2bf(f.w);
    ((uint2*)dst)[i] = o.u;
}

// ---------------------------------------------------------------------------
// Kernel 1: x_proj = X(32768x512) * Wih^T(512x1536) + b_ih  -> bf16 in ws
// Each wave owns a 64x64 register tile (4 M-tiles x 4 N-tiles, 16 v8f accums).
// Per k-step: 8 A b128 loads + 8 B b128 loads + 16 WMMAs, zero conversion VALU.
// ---------------------------------------------------------------------------
#define MT 4
#define NT 4
#define NGROUPS (G3 / (16 * NT))        // 24
#define MGROUPS (MTOT / (16 * MT))      // 512

__global__ __launch_bounds__(256) void xproj_gemm_kernel(
    const unsigned short* __restrict__ Xbf,   // bf16 [32768][512]
    const unsigned short* __restrict__ Wbf,   // bf16 [1536][512]
    const float*          __restrict__ bih,
    unsigned short*       __restrict__ XG) {  // bf16 [32768][1536]

    const int wave = threadIdx.x >> 5;
    const int lane = threadIdx.x & 31;
    const int job  = blockIdx.x * 8 + wave;   // 12288 wave jobs
    const int mg   = job / NGROUPS;
    const int ng   = job % NGROUPS;
    const int nl   = lane & 15;
    const int hi   = lane >> 4;

    const unsigned short* arow[MT];
    const unsigned short* brow[NT];
#pragma unroll
    for (int i = 0; i < MT; ++i)
        arow[i] = Xbf + (size_t)(mg * 64 + i * 16 + nl) * DN;
#pragma unroll
    for (int j = 0; j < NT; ++j)
        brow[j] = Wbf + (size_t)(ng * 64 + j * 16 + nl) * DN;

    v8f acc[MT][NT];
#pragma unroll
    for (int i = 0; i < MT; ++i)
#pragma unroll
        for (int j = 0; j < NT; ++j) acc[i][j] = (v8f){};

#pragma unroll 2
    for (int k0 = 0; k0 < DN; k0 += 32) {
        BF16Frag a[MT];
#pragma unroll
        for (int i = 0; i < MT; ++i) {
            // A layout: 32 contiguous bytes per lane at k = k0 + hi*16
            const uint4* p = (const uint4*)(arow[i] + k0 + hi * 16);
            a[i].q[0] = p[0];
            a[i].q[1] = p[1];
        }
        BF16Frag b[NT];
#pragma unroll
        for (int j = 0; j < NT; ++j) {
            // B layout: 32 contiguous bytes per lane at k = k0 + hi*16
            const uint4* p = (const uint4*)(brow[j] + k0 + hi * 16);
            b[j].q[0] = p[0];
            b[j].q[1] = p[1];
        }
#pragma unroll
        for (int i = 0; i < MT; ++i)
#pragma unroll
            for (int j = 0; j < NT; ++j)
                acc[i][j] = __builtin_amdgcn_wmma_f32_16x16x32_bf16(
                    false, a[i].v, false, b[j].v, (short)0, acc[i][j],
                    false, false);
    }

#pragma unroll
    for (int j = 0; j < NT; ++j) {
        const int g    = ng * 64 + j * 16 + nl;   // C: N = lane&15
        const float bi = bih[g];
#pragma unroll
        for (int i = 0; i < MT; ++i) {
#pragma unroll
            for (int v = 0; v < 8; ++v) {
                int m = mg * 64 + i * 16 + hi * 8 + v;   // C: M = v + hi*8
                XG[(size_t)m * G3 + g] = f2bf(acc[i][j][v] + bi);
            }
        }
    }
}

// ---------------------------------------------------------------------------
// Kernel 2: sequential GRU scan. 16 workgroups; WG `wg` owns hidden units
// [wg*32, wg*32+32) -> 96 rows of bf16 w_hh resident in registers (6 compute
// waves x 16 v16bf fragments = 128 VGPRs). Per step: 16 KB h broadcast via
// L2, 96 WMMAs, gates + mask reset + residual, one global barrier.
// ---------------------------------------------------------------------------
__global__ __launch_bounds__(256) void gru_scan_kernel(
    const float*          __restrict__ X,
    const int*            __restrict__ startIdx,
    const unsigned short* __restrict__ Whhbf,   // bf16 [1536][512]
    const float*          __restrict__ bhh,
    const unsigned short* __restrict__ XG,
    unsigned short* hbuf,      // 2 x [16][512] bf16 double buffer
    int*            cnt,       // [2048] arrive counters (zeroed by init)
    float*          out) {     // [output (B,T,512)] ++ [hidden (B,T,512)]

    __shared__ unsigned short h_lds[BN * HN];   // masked h_prev, bf16, 16 KB
    __shared__ float          hg_buf[96 * 16];  // hg tile exchange, 6 KB
    __shared__ int            sstart[BN];
    __shared__ float          sbias[96];

    const int tid  = threadIdx.x;
    const int lane = tid & 31;
    const int wave = tid >> 5;
    const int wg   = blockIdx.x;                // 0..15
    const int NWG  = 16;

    if (tid < BN) sstart[tid] = startIdx[tid];
    if (tid < 96) {
        int gate = tid >> 5, idx = tid & 31;
        sbias[tid] = bhh[gate * HN + wg * 32 + idx];
    }

    // ---- preload this WG's w_hh slice into registers (pure b128 loads) ----
    BF16Frag bfrag[16];
    if (wave < 6) {
        const int nl = lane & 15, hi = lane >> 4;
        const int rl   = wave * 16 + nl;              // local row 0..95
        const int gate = rl >> 5, idx = rl & 31;
        const unsigned short* wrow =
            Whhbf + (size_t)(gate * HN + wg * 32 + idx) * HN;
#pragma unroll
        for (int kk = 0; kk < 16; ++kk) {
            const uint4* p = (const uint4*)(wrow + kk * 32 + hi * 16);
            bfrag[kk].q[0] = p[0];
            bfrag[kk].q[1] = p[1];
        }
    }
    __syncthreads();

    for (int t = 0; t < TN; ++t) {
        const int par = t & 1;

        // -- stage 1: load h_prev (coherent), apply start-index mask -> LDS
        {
            volatile const unsigned int* hb =
                (volatile const unsigned int*)(hbuf + par * (BN * HN));
            unsigned int* hl = (unsigned int*)h_lds;
#pragma unroll
            for (int i = 0; i < 16; ++i) {
                int idx = tid + i * 256;              // 4096 u32 total
                int b   = idx >> 8;                   // 256 u32 per batch row
                unsigned int v = hb[idx];
                hl[idx] = (sstart[b] < t) ? v : 0u;
            }
        }
        __syncthreads();

        // -- stage 2: hg = h_masked (16x512) * Whh_slice^T (512x96) via WMMA
        if (wave < 6) {
            const int nl = lane & 15, hi = lane >> 4;
            v8f acc = {};
#pragma unroll
            for (int kk = 0; kk < 16; ++kk) {
                BF16Frag a;
                const uint4* q = (const uint4*)
                    (h_lds + nl * HN + kk * 32 + hi * 16);  // 16B-aligned
                a.q[0] = q[0];        // k = kk*32 + hi*16 + 0..7
                a.q[1] = q[1];        // k = kk*32 + hi*16 + 8..15
                acc = __builtin_amdgcn_wmma_f32_16x16x32_bf16(
                          false, a.v, false, bfrag[kk].v, (short)0, acc,
                          false, false);
            }
            const int rl = wave * 16 + nl;
            const float bi = sbias[rl];
#pragma unroll
            for (int v = 0; v < 8; ++v)               // C: batch = v + hi*8
                hg_buf[rl * 16 + hi * 8 + v] = acc[v] + bi;
        }
        __syncthreads();

        // -- stage 3: gates, update, residual output, hidden sequence
        unsigned short* hnext = hbuf + (1 - par) * (BN * HN);
#pragma unroll
        for (int i = 0; i < 2; ++i) {
            int p  = tid + i * 256;                   // 512 (b, jj) pairs
            int b  = p >> 5;
            int jj = p & 31;
            int jg = wg * 32 + jj;                    // global hidden unit
            size_t row    = (size_t)b * TN + t;
            size_t xgbase = row * G3 + wg * 32 + jj;
            float xr = bf2f(XG[xgbase]);
            float xz = bf2f(XG[xgbase + HN]);
            float xn = bf2f(XG[xgbase + 2 * HN]);
            float hr = hg_buf[jj * 16 + b];
            float hz = hg_buf[(32 + jj) * 16 + b];
            float hn = hg_buf[(64 + jj) * 16 + b];
            float hm = bf2f(h_lds[b * HN + jg]);      // masked h_prev
            float r  = 1.0f / (1.0f + __expf(-(xr + hr)));
            float z  = 1.0f / (1.0f + __expf(-(xz + hz)));
            float nn = tanhf(xn + r * hn);
            float hnew = (1.0f - z) * nn + z * hm;
            ((volatile unsigned short*)hnext)[b * HN + jg] = f2bf(hnew);
            size_t o = row * HN + jg;
            out[o] = X[o] + hnew;                                  // residual
            out[(size_t)BN * TN * HN + o] = hnew;                  // hidden seq
        }

        // -- per-step grid-wide barrier (16 co-resident WGs)
        __threadfence();
        __syncthreads();
        if (tid == 0) {
            atomicAdd(&cnt[t], 1);
            while (atomicAdd(&cnt[t], 0) < NWG)
                __builtin_amdgcn_s_sleep(1);
        }
        __syncthreads();
    }
}

// ---------------------------------------------------------------------------
// Host launcher
// ---------------------------------------------------------------------------
extern "C" void kernel_launch(void* const* d_in, const int* in_sizes, int n_in,
                              void* d_out, int out_size, void* d_ws, size_t ws_size,
                              hipStream_t stream) {
    const float* x      = (const float*)d_in[0];   // (16, 2048, 512)
    const int*   start  = (const int*)  d_in[1];   // (16,)
    const float* w_ih   = (const float*)d_in[2];   // (1536, 512)
    const float* w_hh   = (const float*)d_in[3];   // (1536, 512)
    const float* b_ih   = (const float*)d_in[4];   // (1536,)
    const float* b_hh   = (const float*)d_in[5];   // (1536,)
    float*       out    = (float*)d_out;

    // Workspace layout (bytes)
    const size_t XG_BYTES  = (size_t)MTOT * G3 * 2;      // 96 MB  x_proj bf16
    const size_t XBF_BYTES = (size_t)MTOT * DN * 2;      // 32 MB  X bf16
    const size_t W_BYTES   = (size_t)G3 * DN * 2;        // 1.5 MB each weight
    char* wsb = (char*)d_ws;
    unsigned short* XG    = (unsigned short*)(wsb);
    unsigned short* XBF   = (unsigned short*)(wsb + XG_BYTES);
    unsigned short* WIHBF = (unsigned short*)(wsb + XG_BYTES + XBF_BYTES);
    unsigned short* WHHBF = (unsigned short*)(wsb + XG_BYTES + XBF_BYTES + W_BYTES);
    unsigned short* HBUF  = (unsigned short*)(wsb + XG_BYTES + XBF_BYTES + 2 * W_BYTES);
    int*            CNT   = (int*)(wsb + XG_BYTES + XBF_BYTES + 2 * W_BYTES
                                   + 2 * BN * HN * 2);

    // 0) deterministic init of h0 / barrier counters
    init_ws_kernel<<<32, 256, 0, stream>>>((unsigned int*)HBUF, CNT);

    // A) one-pass fp32 -> bf16 conversions
    cvt_bf16_kernel<<<(MTOT * DN / 4 + 255) / 256, 256, 0, stream>>>(x, XBF, MTOT * DN / 4);
    cvt_bf16_kernel<<<(G3 * DN / 4 + 255) / 256, 256, 0, stream>>>(w_ih, WIHBF, G3 * DN / 4);
    cvt_bf16_kernel<<<(G3 * DN / 4 + 255) / 256, 256, 0, stream>>>(w_hh, WHHBF, G3 * DN / 4);

    // 1) x_proj GEMM: 64x64 register tile per wave, 8 waves per block
    const int jobs = MGROUPS * NGROUPS;                  // 12288
    xproj_gemm_kernel<<<jobs / 8, 256, 0, stream>>>(XBF, WIHBF, b_ih, XG);

    // 2) sequential GRU scan: one WG per 32 hidden units
    gru_scan_kernel<<<16, 256, 0, stream>>>(x, start, WHHBF, b_hh, XG,
                                            HBUF, CNT, out);
}
